// TestMMRSModel_3032246911520
// MI455X (gfx1250) — matrix operand order repaired and checked by reading, not compiled, not run on hardware
//
#include <hip/hip_runtime.h>

typedef __attribute__((ext_vector_type(16))) __bf16 v16bf;
typedef __attribute__((ext_vector_type(8)))  __bf16 v8bf;
typedef __attribute__((ext_vector_type(2)))  __bf16 v2bf;
typedef __attribute__((ext_vector_type(8)))  float  v8f;
typedef __attribute__((ext_vector_type(4)))  float  v4f;

#define HID     4096
#define OUTD    8192
#define NROW    2048
#define WORLD   8
#define ASTRIDE 40   // padded LDS row stride (32 data bf16 + 8 pad) = 80B, breaks bank conflicts

// Native f32->bf16 (RNE) pack; clang lowers fptrunc to the gfx1250 cvt ops.
__device__ __forceinline__ unsigned int pack2_bf16(float x, float y) {
  v2bf v;
  v[0] = (__bf16)x;
  v[1] = (__bf16)y;
  return __builtin_bit_cast(unsigned int, v);
}

// ---- gfx1250 async global->LDS path (signature per hipcc diagnostic:
//      param0 = AS1 pointer to vector_size(16) int; param1 = AS3 peer) ----
#if __has_builtin(__builtin_amdgcn_global_load_async_to_lds_b128)
#define HAVE_ASYNC_LDS 1
typedef int v4i_vs __attribute__((vector_size(4 * sizeof(int))));
typedef __attribute__((address_space(1))) v4i_vs* as1_v4i_ptr;
typedef __attribute__((address_space(3))) v4i_vs* as3_v4i_ptr;
__device__ __forceinline__ void async_copy16(void* lds, const void* g) {
  __builtin_amdgcn_global_load_async_to_lds_b128(
      (as1_v4i_ptr)(g), (as3_v4i_ptr)(lds), /*offset=*/0, /*cpol=*/0);
}
#else
#define HAVE_ASYNC_LDS 0
__device__ __forceinline__ void async_copy16(void* lds, const void* g) {
  *(uint4*)lds = *(const uint4*)g;
}
#endif

__device__ __forceinline__ void async_wait0() {
#if HAVE_ASYNC_LDS
#if __has_builtin(__builtin_amdgcn_s_wait_asynccnt)
  __builtin_amdgcn_s_wait_asynccnt(0);
#else
  asm volatile("s_wait_asynccnt 0" ::: "memory");
#endif
#endif
}

// ---------------------------------------------------------------------------
// Pass 1: A_red[n][h] = sum_w hidden[w][n][h]  (f32 accumulate), store bf16.
// Pure bandwidth: 256MB read, 16MB write.
// ---------------------------------------------------------------------------
__global__ __launch_bounds__(256) void reduce_ws_bf16(const float* __restrict__ hs,
                                                      unsigned short* __restrict__ ared) {
  const size_t q = ((size_t)blockIdx.x * 256 + threadIdx.x) * 4;  // f32 index in [NROW*HID]
  v4f s = {0.f, 0.f, 0.f, 0.f};
#pragma unroll
  for (int w = 0; w < WORLD; ++w) {
    v4f v = *(const v4f*)(hs + (size_t)w * NROW * HID + q);
    s += v;
  }
  *(uint2*)(ared + q) = make_uint2(pack2_bf16(s[0], s[1]), pack2_bf16(s[2], s[3]));
}

// ---------------------------------------------------------------------------
// Pass 2: C[2048][8192] = A_red[2048][4096](bf16) @ W[8192][4096](f32->bf16)^T
// 128x128 block tile, K-step 32, 8 waves as 2(M)x4(N), 4x2 WMMAs per wave.
// A tile: async global->LDS (already bf16).  B tile: load f32, cvt, ds_store.
// ---------------------------------------------------------------------------
__global__ __launch_bounds__(256) void mmrs_gemm_bf16(const unsigned short* __restrict__ A,
                                                      const float* __restrict__ Wt,
                                                      float* __restrict__ C) {
  __shared__ __align__(16) unsigned short As[128 * ASTRIDE];
  __shared__ __align__(16) unsigned short Bs[128 * ASTRIDE];

  const int tid  = threadIdx.x;
  const int lane = tid & 31;
  const int l16  = lane & 15;
  const int lhi  = lane >> 4;      // 0: low half, 1: high half of wave
  const int wv   = tid >> 5;       // wave id 0..7
  const int wm   = wv & 1;         // 2 wave-rows  -> 64 M-rows each
  const int wn   = wv >> 1;        // 4 wave-cols  -> 32 N-cols each

  const int m0 = blockIdx.y * 128;
  const int n0 = blockIdx.x * 128;

  // A staging: 512 x 16B chunks; thread covers chunk tid and tid+256
  const int ar = tid >> 2;              // row 0..63 (+64 for second chunk)
  const int ac = (tid & 3) * 8;         // bf16 column 0/8/16/24
  const unsigned short* aG0 = A + (size_t)(m0 + ar) * HID + ac;
  const unsigned short* aG1 = A + (size_t)(m0 + ar + 64) * HID + ac;
  unsigned short* aL0 = &As[ar * ASTRIDE + ac];
  unsigned short* aL1 = &As[(ar + 64) * ASTRIDE + ac];

  // B staging: 1024 x float4 slots; thread covers 4 rows (tid>>3 + i*32)
  const int br = tid >> 3;              // row 0..31 (+32*i)
  const int bc = (tid & 7) * 4;         // f32 column
  const float* bG = Wt + (size_t)(n0 + br) * HID + bc;

  const v8f zacc = {0.f, 0.f, 0.f, 0.f, 0.f, 0.f, 0.f, 0.f};
  v8f acc[4][2];
#pragma unroll
  for (int i = 0; i < 4; ++i)
#pragma unroll
    for (int j = 0; j < 2; ++j) acc[i][j] = zacc;

  for (int k0 = 0; k0 < HID; k0 += 32) {
    // B: global -> regs early (overlaps previous iteration's WMMAs)
    v4f bR[4];
#pragma unroll
    for (int i = 0; i < 4; ++i)
      bR[i] = *(const v4f*)(bG + (size_t)i * 32 * HID + k0);

    __syncthreads();   // previous tile's LDS reads done -> safe to overwrite

    // A: async global->LDS, no VGPR round-trip (ASYNCcnt-tracked)
    async_copy16(aL0, aG0 + k0);
    async_copy16(aL1, aG1 + k0);

    // B: convert f32 -> bf16, ds_store (runs while async A is in flight)
#pragma unroll
    for (int i = 0; i < 4; ++i)
      *(uint2*)&Bs[(br + i * 32) * ASTRIDE + bc] =
          make_uint2(pack2_bf16(bR[i][0], bR[i][1]), pack2_bf16(bR[i][2], bR[i][3]));

    async_wait0();     // this wave's async A writes landed in LDS
    __syncthreads();   // all waves' A+B visible

    if (k0 + 32 < HID) {  // warm next K-step (global_prefetch_b8)
      __builtin_prefetch(aG0 + k0 + 32, 0, 0);
      __builtin_prefetch(bG + k0 + 32, 0, 0);
    }

    // A fragments: 16x32 bf16. lanes 0-15: M=lane, K 0-7 & 16-23;
    // lanes 16-31: M=lane-16, K 8-15 & 24-31.  Two ds_load_b128 each.
    v16bf afr[4];
#pragma unroll
    for (int i = 0; i < 4; ++i) {
      const int row = wm * 64 + i * 16 + l16;
      v8bf lo = *(const v8bf*)&As[row * ASTRIDE + lhi * 8];
      v8bf hi = *(const v8bf*)&As[row * ASTRIDE + lhi * 8 + 16];
      afr[i] = __builtin_shufflevector(lo, hi, 0,1,2,3,4,5,6,7,8,9,10,11,12,13,14,15);
    }
    // B fragments: 32x16 bf16. lanes 0-15: N=lane, K 0-7 & 16-23;
    // lanes 16-31: N=lane-16, K 8-15 & 24-31.  Same K order as A.
    v16bf bfr[2];
#pragma unroll
    for (int j = 0; j < 2; ++j) {
      const int row = wn * 32 + j * 16 + l16;
      v8bf lo = *(const v8bf*)&Bs[row * ASTRIDE + lhi * 8];
      v8bf hi = *(const v8bf*)&Bs[row * ASTRIDE + lhi * 8 + 16];
      bfr[j] = __builtin_shufflevector(lo, hi, 0,1,2,3,4,5,6,7,8,9,10,11,12,13,14,15);
    }

#pragma unroll
    for (int i = 0; i < 4; ++i)
#pragma unroll
      for (int j = 0; j < 2; ++j)
        acc[i][j] = __builtin_amdgcn_wmma_f32_16x16x32_bf16(
            /*neg_a=*/false, afr[i], /*neg_b=*/false, bfr[j],
            /*c_mod=*/(short)0, acc[i][j], /*reuse_a=*/false, /*reuse_b=*/false);
  }

  // C layout: VGPR e, lanes 0-15: M=e, N=lane; lanes 16-31: M=e+8, N=lane-16
#pragma unroll
  for (int i = 0; i < 4; ++i) {
    const int rbase = m0 + wm * 64 + i * 16 + lhi * 8;
#pragma unroll
    for (int j = 0; j < 2; ++j) {
      const int col = n0 + wn * 32 + j * 16 + l16;
#pragma unroll
      for (int e = 0; e < 8; ++e)
        C[(size_t)(rbase + e) * OUTD + col] = acc[i][j][e];
    }
  }
}

// ---------------------------------------------------------------------------
extern "C" void kernel_launch(void* const* d_in, const int* in_sizes, int n_in,
                              void* d_out, int out_size, void* d_ws, size_t ws_size,
                              hipStream_t stream) {
  (void)in_sizes; (void)n_in; (void)out_size; (void)ws_size;
  const float* hs = (const float*)d_in[0];        // [8, 2048, 4096] f32
  const float* gp = (const float*)d_in[1];        // [8192, 4096]    f32
  float* out = (float*)d_out;                     // [2048, 8192]    f32
  unsigned short* ared = (unsigned short*)d_ws;   // [2048, 4096]    bf16 (16 MB)

  reduce_ws_bf16<<<(NROW * HID / 4) / 256, 256, 0, stream>>>(hs, ared);

  dim3 grid(OUTD / 128, NROW / 128);              // 64 x 16 = 1024 blocks
  mmrs_gemm_bf16<<<grid, dim3(256, 1, 1), 0, stream>>>(ared, gp, out);
}
